// STDCN_with_GRU_9680856285767
// MI455X (gfx1250) — matrix operand order repaired and checked by reading, not compiled, not run on hardware
//
#include <hip/hip_runtime.h>
#include <math.h>

// Dims: B=512, C=3, N=128 (in_dim), F=256 (seq), HID=128, 3H=384
typedef __bf16 bf16x16 __attribute__((ext_vector_type(16)));
typedef float floatx8 __attribute__((ext_vector_type(8)));

#define NB 512
#define NC 3
#define NN 128
#define NF 256
#define NH 128
#define G3 384
#define XT_STRIDE 136   // padded row stride (bf16) -> 272B rows, 16B aligned, bank-rotating

// -------------------------------------------------------------------------
// Kernel 0: dinv[b,c,n] = rsqrt(1 + sum_{m!=n} H[b,c,n,m])
// (sum whole row, subtract diagonal afterwards -> branch-free inner loop)
// -------------------------------------------------------------------------
__global__ __launch_bounds__(256) void degree_kernel(const float* __restrict__ H,
                                                     float* __restrict__ dinv) {
    int idx = blockIdx.x * 256 + threadIdx.x;          // B*C*N = 196608
    int n = idx & 127;
    const float* row = H + (size_t)idx * NN;
    float s = 0.0f;
    for (int m = 0; m < NN; ++m) s += row[m];
    s = 1.0f + s - row[n];
    dinv[idx] = rsqrtf(s);
}

// -------------------------------------------------------------------------
// Kernel 1: GRU weights f32 -> bf16  (layout: wih_f | whh_f | wih_b | whh_b)
// -------------------------------------------------------------------------
__global__ __launch_bounds__(256) void cvt_w_kernel(const float* __restrict__ a,
                                                    const float* __restrict__ b,
                                                    const float* __restrict__ c,
                                                    const float* __restrict__ d,
                                                    __bf16* __restrict__ dst) {
    int i = blockIdx.x * 256 + threadIdx.x;            // 4 * 384*128 = 196608
    int which = i / (G3 * NH);
    int j = i - which * (G3 * NH);
    const float* src = (which == 0) ? a : (which == 1) ? b : (which == 2) ? c : d;
    dst[i] = (__bf16)src[j];
}

// -------------------------------------------------------------------------
// Kernel 2: normalized GCN conv + leaky_relu + soft-sum over channels.
// X half tile stored TRANSPOSED in LDS (XT[f][k], padded stride) so the WMMA
// B-operand is a per-lane contiguous 32B run -> 2x ds_load_b128 per chunk.
// Output written time-major for the GRU: xs[t][b][n] (bf16).
// Grid: B * 8 ntiles * 2 fhalves = 8192 blocks, 256 threads (8 waves).
// -------------------------------------------------------------------------
__global__ __launch_bounds__(256) void gcn_kernel(const float* __restrict__ X,
                                                  const float* __restrict__ H,
                                                  const float* __restrict__ dinv,
                                                  const float* __restrict__ weight,
                                                  const float* __restrict__ soft,
                                                  __bf16* __restrict__ xs) {
    __shared__ __bf16 XT[128 * XT_STRIDE]; // X[b][:, fh*128+f] at XT[f*136 + m] (34 KB)
    __shared__ __bf16 As[16 * NN];         // scaled Hn tile, current channel (4 KB)

    const int bid = blockIdx.x;
    const int b = bid >> 4;
    const int ntile = (bid >> 1) & 7;
    const int fh = bid & 1;
    const int tid = threadIdx.x;
    const int lane = tid & 31;
    const int wv = tid >> 5;
    const int half = lane >> 4;
    const int mrow = lane & 15;
    const float w10 = weight[0];

    // coalesced global read (f fastest), transposed LDS write
    for (int i = tid; i < NN * 128; i += 256) {
        int m = i >> 7, f = i & 127;
        XT[f * XT_STRIDE + m] =
            (__bf16)X[(size_t)b * NN * NF + (size_t)m * NF + fh * 128 + f];
    }
    __syncthreads();

    const int fcol = wv * 16 + (lane & 15);
    float vout[8] = {0.f, 0.f, 0.f, 0.f, 0.f, 0.f, 0.f, 0.f};

    for (int c = 0; c < NC; ++c) {
        const float* Hbase = H + ((size_t)(b * NC + c) * NN) * NN;
        const float* dv = dinv + (b * NC + c) * NN;
        for (int i = tid; i < 16 * NN; i += 256) {
            int nl = i >> 7, mc = i & 127;
            int ng = ntile * 16 + nl;
            float hv = Hbase[(size_t)ng * NN + mc];    // unconditional load
            hv = (mc == ng) ? 1.0f : hv;               // blend -> v_cndmask
            As[i] = (__bf16)(hv * dv[ng] * dv[mc]);
        }
        __syncthreads();

        floatx8 acc = {0.f, 0.f, 0.f, 0.f, 0.f, 0.f, 0.f, 0.f};
#pragma unroll
        for (int kc = 0; kc < 4; ++kc) {
            bf16x16 a, bm;
#pragma unroll
            for (int i = 0; i < 16; ++i) {          // A: one contiguous 16-elem run
                int k = kc * 32 + half * 16 + i;
                a[i] = As[mrow * NN + k];
            }
#pragma unroll
            for (int i = 0; i < 16; ++i) {          // B: one contiguous 16-elem run
                int k = kc * 32 + half * 16 + i;
                bm[i] = XT[fcol * XT_STRIDE + k];
            }
            acc = __builtin_amdgcn_wmma_f32_16x16x32_bf16(false, a, false, bm,
                                                          (short)0, acc, false, false);
        }
        float sc = soft[c];
#pragma unroll
        for (int r = 0; r < 8; ++r) {
            float v = acc[r] * w10;
            v = (v > 0.0f) ? v : 0.01f * v;   // leaky_relu(0.01)
            vout[r] += v * sc;
        }
        __syncthreads();
    }

#pragma unroll
    for (int r = 0; r < 8; ++r) {
        int m = ntile * 16 + r + half * 8;
        int fg = fh * 128 + fcol;
        xs[(size_t)fg * (NB * NN) + (size_t)b * NN + m] = (__bf16)vout[r];
    }
}

// -------------------------------------------------------------------------
// Kernel 3: bidirectional GRU. Wave w owns gate-column tiles {w, w+8, w+16}:
// the r/z/n accumulators for a given (batch,row) land in the same lane/elem,
// so all gate math is register-resident. Only the 4 KB bf16 h copy touches
// LDS (next step's WMMA A operand). h_old lives in 8 regs/lane (C layout).
// Grid: 64 blocks = 32 batch-blocks (16 rows) x 2 directions, 256 threads.
// -------------------------------------------------------------------------
__global__ __launch_bounds__(256) void gru_kernel(const __bf16* __restrict__ xs,
                                                  const __bf16* __restrict__ wbf,
                                                  const float* __restrict__ b_ih_f,
                                                  const float* __restrict__ b_hh_f,
                                                  const float* __restrict__ b_ih_b,
                                                  const float* __restrict__ b_hh_b,
                                                  float* __restrict__ out) {
    __shared__ __bf16 hb[16 * NH];   // 4 KB: h in bf16 for the A operand

    const int dir = blockIdx.x >> 5;
    const int b0 = (blockIdx.x & 31) * 16;
    const int tid = threadIdx.x;
    const int lane = tid & 31;
    const int wv = tid >> 5;
    const int half = lane >> 4;
    const int mrow = lane & 15;
    const int ncol = lane & 15;
    const int jcol = wv * 16 + ncol;           // hidden column this lane owns

    const __bf16* wih = wbf + (dir ? 2 * G3 * NH : 0);
    const __bf16* whh = wih + G3 * NH;
    const float* bih = dir ? b_ih_b : b_ih_f;
    const float* bhh = dir ? b_hh_b : b_hh_f;

    // warm L2 with this direction's weights
    __builtin_prefetch(wih + (size_t)jcol * NH, 0, 1);
    __builtin_prefetch(whh + (size_t)jcol * NH, 0, 1);

    // time-invariant per-lane biases
    const float bi_r = bih[jcol], bi_z = bih[NH + jcol], bi_n = bih[2 * NH + jcol];
    const float bh_r = bhh[jcol], bh_z = bhh[NH + jcol], bh_n = bhh[2 * NH + jcol];

    float hreg[8] = {0.f, 0.f, 0.f, 0.f, 0.f, 0.f, 0.f, 0.f};
    for (int i = tid; i < 16 * NH; i += 256) hb[i] = (__bf16)0.0f;
    __syncthreads();

    for (int ti = 0; ti < NF; ++ti) {
        const int t = dir ? (NF - 1 - ti) : ti;   // read & write index coincide
        const __bf16* xrow = xs + (size_t)t * (NB * NN) + (size_t)(b0 + mrow) * NN;

        floatx8 acc_i[3], acc_h[3];
#pragma unroll
        for (int g = 0; g < 3; ++g) {
            acc_i[g] = (floatx8){0.f, 0.f, 0.f, 0.f, 0.f, 0.f, 0.f, 0.f};
            acc_h[g] = (floatx8){0.f, 0.f, 0.f, 0.f, 0.f, 0.f, 0.f, 0.f};
        }

#pragma unroll
        for (int kc = 0; kc < 4; ++kc) {
            bf16x16 ax, am;
#pragma unroll
            for (int i = 0; i < 16; ++i) {        // A: one contiguous 16-elem run
                int k = kc * 32 + half * 16 + i;
                ax[i] = xrow[k];
                am[i] = hb[mrow * NH + k];
            }
#pragma unroll
            for (int g = 0; g < 3; ++g) {         // gate blocks r/z/n
                const int col = g * NH + jcol;
                bf16x16 bi, bh2;
#pragma unroll
                for (int i = 0; i < 16; ++i) {    // B: contiguous 32B run per lane
                    int k = kc * 32 + half * 16 + i;
                    bi[i] = wih[(size_t)col * NH + k];
                    bh2[i] = whh[(size_t)col * NH + k];
                }
                acc_i[g] = __builtin_amdgcn_wmma_f32_16x16x32_bf16(
                    false, ax, false, bi, (short)0, acc_i[g], false, false);
                acc_h[g] = __builtin_amdgcn_wmma_f32_16x16x32_bf16(
                    false, am, false, bh2, (short)0, acc_h[g], false, false);
            }
        }
        __syncthreads();   // all waves done reading old hb

#pragma unroll
        for (int r = 0; r < 8; ++r) {
            const int m = r + half * 8;
            float rg = 1.0f / (1.0f + __expf(-(acc_i[0][r] + bi_r + acc_h[0][r] + bh_r)));
            float zg = 1.0f / (1.0f + __expf(-(acc_i[1][r] + bi_z + acc_h[1][r] + bh_z)));
            float ng = tanhf(acc_i[2][r] + bi_n + rg * (acc_h[2][r] + bh_n));
            float hnew = (1.0f - zg) * ng + zg * hreg[r];
            hreg[r] = hnew;
            hb[m * NH + jcol] = (__bf16)hnew;
            out[(size_t)(b0 + m) * (2 * NH * NF) + (size_t)(dir * NH + jcol) * NF + t] = hnew;
        }
        __syncthreads();   // new hb visible before next step's reads
    }
}

// -------------------------------------------------------------------------
// Kernel 4/5: BatchNorm1d(F) in training mode over axes (batch, 2H).
// out holds y transposed to [B, 2H, F]; stats are per f = flat_index & 255.
// -------------------------------------------------------------------------
__global__ __launch_bounds__(256) void bn_stats_kernel(const float* __restrict__ y,
                                                       float* __restrict__ stats) {
    __shared__ float rs[256], rq[256];
    const int f = blockIdx.x;
    const int tid = threadIdx.x;
    float s = 0.0f, q = 0.0f;
    for (int i = tid; i < NB * 2 * NH; i += 256) {
        float v = y[(size_t)i * NF + f];
        s += v;
        q += v * v;
    }
    rs[tid] = s;
    rq[tid] = q;
    __syncthreads();
    for (int off = 128; off > 0; off >>= 1) {
        if (tid < off) { rs[tid] += rs[tid + off]; rq[tid] += rq[tid + off]; }
        __syncthreads();
    }
    if (tid == 0) {
        const float inv = 1.0f / (float)(NB * 2 * NH);
        float mean = rs[0] * inv;
        float var = rq[0] * inv - mean * mean;
        stats[f] = mean;
        stats[NF + f] = var;
    }
}

__global__ __launch_bounds__(256) void bn_apply_kernel(float* __restrict__ y,
                                                       const float* __restrict__ stats,
                                                       const float* __restrict__ gamma,
                                                       const float* __restrict__ beta) {
    size_t i = (size_t)blockIdx.x * 256 + threadIdx.x;
    int f = (int)(i & (NF - 1));
    float mean = stats[f];
    float var = stats[NF + f];
    y[i] = gamma[f] * (y[i] - mean) * rsqrtf(var + 1e-5f) + beta[f];
}

// -------------------------------------------------------------------------
extern "C" void kernel_launch(void* const* d_in, const int* in_sizes, int n_in,
                              void* d_out, int out_size, void* d_ws, size_t ws_size,
                              hipStream_t stream) {
    const float* X      = (const float*)d_in[0];
    const float* H      = (const float*)d_in[1];
    const float* weight = (const float*)d_in[2];
    const float* soft   = (const float*)d_in[3];
    const float* w_ih_f = (const float*)d_in[4];
    const float* w_hh_f = (const float*)d_in[5];
    const float* b_ih_f = (const float*)d_in[6];
    const float* b_hh_f = (const float*)d_in[7];
    const float* w_ih_b = (const float*)d_in[8];
    const float* w_hh_b = (const float*)d_in[9];
    const float* b_ih_b = (const float*)d_in[10];
    const float* b_hh_b = (const float*)d_in[11];
    const float* gamma  = (const float*)d_in[12];
    const float* beta   = (const float*)d_in[13];
    float* out = (float*)d_out;

    char* ws = (char*)d_ws;
    const size_t dinv_bytes = (size_t)NB * NC * NN * 4;          // 786432
    const size_t xs_bytes   = (size_t)NF * NB * NN * 2;          // 33554432
    const size_t wbf_bytes  = (size_t)4 * G3 * NH * 2;           // 393216
    float*  dinv  = (float*)ws;
    __bf16* xs    = (__bf16*)(ws + dinv_bytes);
    __bf16* wbf   = (__bf16*)(ws + dinv_bytes + xs_bytes);
    float*  stats = (float*)(ws + dinv_bytes + xs_bytes + wbf_bytes);
    (void)ws_size; (void)in_sizes; (void)n_in; (void)out_size;

    // 0) degree / dinv
    degree_kernel<<<(NB * NC * NN) / 256, 256, 0, stream>>>(H, dinv);
    // 1) weights -> bf16
    cvt_w_kernel<<<(4 * G3 * NH) / 256, 256, 0, stream>>>(w_ih_f, w_hh_f, w_ih_b, w_hh_b, wbf);
    // 2) GCN conv -> xs[t][b][n] bf16
    gcn_kernel<<<NB * 16, 256, 0, stream>>>(X, H, dinv, weight, soft, xs);
    // 3) bidirectional GRU (sequential over 256 steps inside each block)
    gru_kernel<<<64, 256, 0, stream>>>(xs, wbf, b_ih_f, b_hh_f, b_ih_b, b_hh_b, out);
    // 4) BN stats per F
    bn_stats_kernel<<<NF, 256, 0, stream>>>(out, stats);
    // 5) BN apply in place
    bn_apply_kernel<<<(NB * 2 * NH * NF) / 256, 256, 0, stream>>>(out, stats, gamma, beta);
}
